// TransformerDSOAgent_UseLayerLearning_60859686584840
// MI455X (gfx1250) — matrix operand order repaired and checked by reading, not compiled, not run on hardware
//
#include <hip/hip_runtime.h>
#include <hip/hip_bf16.h>

typedef __attribute__((ext_vector_type(8)))  __bf16 v8bf;
typedef __attribute__((ext_vector_type(16))) __bf16 v16bf;
typedef __attribute__((ext_vector_type(8)))  float  v8f;

#define HDIM  1024
#define PDIM  128
#define INDIM 263
#define K0PAD 416   // 263 + 128 -> padded to multiple of 32
#define K1PAD 256   // 128 + 128
#define XS0   432   // LDS row stride (bf16) for layer-0 A, 16B-aligned rows
#define XS1   272   // LDS row stride for layer-1 A
#define AS    144   // LDS row stride for activation chunk buffer
#define GS    136   // LDS row stride (floats) for softmax buffer
#define MROWS 32    // batch rows per workgroup (2 M-tiles) -> halves L2 weight traffic

__device__ __forceinline__ unsigned short f2bf(float f) {
  unsigned int u = __float_as_uint(f);
  u += 0x7FFFu + ((u >> 16) & 1u);   // round-to-nearest-even
  return (unsigned short)(u >> 16);
}

__device__ __forceinline__ float sigmf(float x) {
  return 1.0f / (1.0f + __expf(-x));
}

__device__ __forceinline__ v8f wmma_bf16(v16bf a, v16bf b, v8f c) {
  // v_wmma_f32_16x16x32_bf16: (neg_a, A, neg_b, B, c_mod, C, reuse_a, reuse_b)
  return __builtin_amdgcn_wmma_f32_16x16x32_bf16(false, a, false, b, (short)0, c,
                                                 false, false);
}

// A-matrix 16x32 bf16 fragment (ISA 7.12.2): lane<16 holds row=rowOff+lane,
// K {k0..k0+7, k0+16..k0+23}; lane>=16 holds row=rowOff+lane-16, K {+8..+15, +24..+31}.
__device__ __forceinline__ v16bf load_a_tile(const unsigned short* base, int stride,
                                             int k0, int lane, int rowOff) {
  int r = rowOff + (lane & 15);
  int koff = k0 + ((lane >> 4) << 3);
  const __bf16* p = (const __bf16*)(base + r * stride + koff);
  v8bf lo = *(const v8bf*)p;          // K k..k+7
  v8bf hi = *(const v8bf*)(p + 16);   // K k+16..k+23
  return __builtin_shufflevector(lo, hi, 0,1,2,3,4,5,6,7,8,9,10,11,12,13,14,15);
}

// B-matrix 32x16 bf16 fragment: lane<16 holds col=lane, K {k0..k0+7, k0+16..k0+23};
// lane>=16 holds col=lane-16, K {+8..+15, +24..+31}. B[k][n] = W[row0+n][k] (weights are [out,in]).
__device__ __forceinline__ v16bf load_b_tile(const unsigned short* W, int stride,
                                             int row0, int k0, int lane) {
  int r = row0 + (lane & 15);
  int koff = k0 + ((lane >> 4) << 3);
  const __bf16* p = (const __bf16*)(W + (size_t)r * stride + koff);
  v8bf lo = *(const v8bf*)p;
  v8bf hi = *(const v8bf*)(p + 16);
  return __builtin_shufflevector(lo, hi, 0,1,2,3,4,5,6,7,8,9,10,11,12,13,14,15);
}

// One projected-LSTM layer for a 32-row block. 8 waves; per 128-unit chunk each
// wave owns 16 hidden units (4 gates x 2 M-tiles = 8 accumulators; each B tile
// feeds both M-tiles) and h-projection columns 16w..16w+15 for both M-tiles.
template<int KPAD, int XS>
__device__ __forceinline__ void run_layer(
    const unsigned short* xlds,              // LDS A matrix [32][XS] bf16
    const unsigned short* __restrict__ Wg,   // bf16 gate weights [4096][KPAD]
    const unsigned short* __restrict__ Wr,   // bf16 proj weights [128][1024]
    const float* __restrict__ bias,          // fused bias [4096]
    const float* __restrict__ c_in,          // c-state [N][1024] for this layer
    float* __restrict__ c_out,               // cn output [N][1024]
    unsigned short* a_buf,                   // LDS [32][AS] bf16 activation chunk
    v8f (&hacc)[2], int lane, int w, int rowBase)
{
  const int nl = lane & 15;
  const int mhalf = (lane >> 4) << 3;
#pragma unroll 1
  for (int hc = 0; hc < HDIM; hc += 128) {
    const int u0 = hc + 16 * w;
    float b4[4];
    b4[0] = bias[u0 + nl];
    b4[1] = bias[HDIM + u0 + nl];
    b4[2] = bias[2 * HDIM + u0 + nl];
    b4[3] = bias[3 * HDIM + u0 + nl];
    v8f acc[2][4];
#pragma unroll
    for (int g = 0; g < 4; ++g)
#pragma unroll
      for (int e = 0; e < 8; ++e) { acc[0][g][e] = b4[g]; acc[1][g][e] = b4[g]; }

#pragma unroll 2
    for (int k0 = 0; k0 < KPAD; k0 += 32) {
      v16bf A0 = load_a_tile(xlds, XS, k0, lane, 0);
      v16bf A1 = load_a_tile(xlds, XS, k0, lane, 16);
#pragma unroll
      for (int g = 0; g < 4; ++g) {
        v16bf B = load_b_tile(Wg, KPAD, g * HDIM + u0, k0, lane);
        acc[0][g] = wmma_bf16(A0, B, acc[0][g]);
        acc[1][g] = wmma_bf16(A1, B, acc[1][g]);
      }
    }

    // Elementwise LSTM update entirely in C/D layout: element e -> (m, n)
#pragma unroll
    for (int t = 0; t < 2; ++t) {
#pragma unroll
      for (int e = 0; e < 8; ++e) {
        int m = 16 * t + e + mhalf;
        int u = u0 + nl;
        size_t idx = (size_t)(rowBase + m) * HDIM + u;
        float c_old = c_in[idx];
        float cnew = sigmf(acc[t][1][e]) * c_old + sigmf(acc[t][0][e]) * tanhf(acc[t][2][e]);
        c_out[idx] = cnew;
        a_buf[m * AS + 16 * w + nl] = f2bf(sigmf(acc[t][3][e]) * tanhf(cnew));
      }
    }
    __syncthreads();

    // h += a_chunk[32x128] @ Wr[:, hc:hc+128]^T  (wave w: proj cols 16w..16w+15)
#pragma unroll
    for (int k0 = 0; k0 < 128; k0 += 32) {
      v16bf A0 = load_a_tile(a_buf, AS, k0, lane, 0);
      v16bf A1 = load_a_tile(a_buf, AS, k0, lane, 16);
      v16bf B = load_b_tile(Wr, HDIM, 16 * w, hc + k0, lane);
      hacc[0] = wmma_bf16(A0, B, hacc[0]);
      hacc[1] = wmma_bf16(A1, B, hacc[1]);
    }
    __syncthreads();
  }
}

__global__ __launch_bounds__(256)
void fused_lstm2_softmax(
    const float* __restrict__ input, const float* __restrict__ hidden,
    const float* __restrict__ hidden_lstm, const int* __restrict__ curlen,
    const unsigned short* __restrict__ Wg0, const unsigned short* __restrict__ Wg1,
    const unsigned short* __restrict__ Wr0, const unsigned short* __restrict__ Wr1,
    const float* __restrict__ bb0, const float* __restrict__ bb1,
    float* __restrict__ out, float* __restrict__ cn0, float* __restrict__ cn1,
    float* __restrict__ hn0, float* __restrict__ hn1, int N)
{
  __shared__ unsigned short xh0[MROWS * XS0];
  __shared__ unsigned short xh1[MROWS * XS1];
  __shared__ unsigned short abuf[MROWS * AS];
  __shared__ float gbuf[MROWS * GS];

  const int tid = threadIdx.x;
  const int lane = tid & 31;
  const int w = tid >> 5;
  const int rowBase = blockIdx.x * MROWS;
  const int mhalf = (lane >> 4) << 3;

  // Stage layer-0 A matrix: [x (263) | h_lstm0 (128) | zero pad]
  for (int i = tid; i < MROWS * XS0; i += 256) {
    int m = i / XS0, c = i - m * XS0;
    float v = 0.0f;
    if (c < INDIM)              v = input[(size_t)(rowBase + m) * INDIM + c];
    else if (c < INDIM + PDIM)  v = hidden_lstm[(size_t)(rowBase + m) * PDIM + (c - INDIM)];
    xh0[i] = f2bf(v);
  }
  __syncthreads();

  v8f h0[2];
#pragma unroll
  for (int t = 0; t < 2; ++t)
#pragma unroll
    for (int e = 0; e < 8; ++e) h0[t][e] = 0.0f;
  run_layer<K0PAD, XS0>(xh0, Wg0, Wr0, bb0, hidden, cn0, abuf, h0, lane, w, rowBase);

  // h0 -> hn[0] (fp32) and into layer-1 A matrix (bf16, transposed via LDS)
#pragma unroll
  for (int t = 0; t < 2; ++t) {
#pragma unroll
    for (int e = 0; e < 8; ++e) {
      int m = 16 * t + e + mhalf;
      int p = 16 * w + (lane & 15);
      float v = h0[t][e];
      hn0[(size_t)(rowBase + m) * PDIM + p] = v;
      xh1[m * XS1 + p] = f2bf(v);
    }
  }
  // h_lstm1 into K columns [128,256)
  for (int i = tid; i < MROWS * PDIM; i += 256) {
    int m = i >> 7, c = i & 127;
    xh1[m * XS1 + PDIM + c] =
        f2bf(hidden_lstm[(size_t)N * PDIM + (size_t)(rowBase + m) * PDIM + c]);
  }
  __syncthreads();

  v8f h1[2];
#pragma unroll
  for (int t = 0; t < 2; ++t)
#pragma unroll
    for (int e = 0; e < 8; ++e) h1[t][e] = 0.0f;
  run_layer<K1PAD, XS1>(xh1, Wg1, Wr1, bb1, hidden + (size_t)N * HDIM, cn1,
                        abuf, h1, lane, w, rowBase);

  // h1 -> hn[1] and LDS for softmax
#pragma unroll
  for (int t = 0; t < 2; ++t) {
#pragma unroll
    for (int e = 0; e < 8; ++e) {
      int m = 16 * t + e + mhalf;
      int p = 16 * w + (lane & 15);
      float v = h1[t][e];
      hn1[(size_t)(rowBase + m) * PDIM + p] = v;
      gbuf[m * GS + p] = v;
    }
  }
  __syncthreads();

  // softmax(h1 + prior) over 128 cols; wave w handles rows 4w..4w+3
  int cl = *curlen;
  float dA = (float)(64 - cl);
  float pA = (cl < 64) ? -(dA * dA) * (1.0f / 16.0f) : 0.0f;
  float dB = (float)(cl - 64);
  float pB = (cl > 64) ? -(dB * dB) * (1.0f / 16.0f) : 0.0f;
#pragma unroll
  for (int rr = 0; rr < 4; ++rr) {
    int row = 4 * w + rr;
    float x[4];
    float mx = -3.4e38f;
#pragma unroll
    for (int q = 0; q < 4; ++q) {
      int c = lane + 32 * q;
      float pr = (c < 32) ? pA : ((c >= 96) ? pB : 0.0f);
      x[q] = gbuf[row * GS + c] + pr;
      mx = fmaxf(mx, x[q]);
    }
#pragma unroll
    for (int off = 16; off >= 1; off >>= 1) mx = fmaxf(mx, __shfl_xor(mx, off, 32));
    float s = 0.0f;
#pragma unroll
    for (int q = 0; q < 4; ++q) { x[q] = __expf(x[q] - mx); s += x[q]; }
#pragma unroll
    for (int off = 16; off >= 1; off >>= 1) s += __shfl_xor(s, off, 32);
    float inv = 1.0f / s;
#pragma unroll
    for (int q = 0; q < 4; ++q)
      out[(size_t)(rowBase + row) * PDIM + lane + 32 * q] = x[q] * inv;
  }
}

// Pack fp32 weights -> bf16 layouts in d_ws; fuse biases. Runs every launch (deterministic).
__global__ void prep_weights(
    const float* __restrict__ W_ih0, const float* __restrict__ W_hh0,
    const float* __restrict__ b_ih0, const float* __restrict__ b_hh0,
    const float* __restrict__ W_hr0,
    const float* __restrict__ W_ih1, const float* __restrict__ W_hh1,
    const float* __restrict__ b_ih1, const float* __restrict__ b_hh1,
    const float* __restrict__ W_hr1,
    unsigned short* __restrict__ Wg0, unsigned short* __restrict__ Wg1,
    unsigned short* __restrict__ Wr0, unsigned short* __restrict__ Wr1,
    float* __restrict__ bb0, float* __restrict__ bb1)
{
  const int S0 = 4096 * K0PAD;
  const int S1 = 4096 * K1PAD;
  const int S2 = PDIM * HDIM;
  int idx = blockIdx.x * 256 + threadIdx.x;
  if (idx < S0) {
    int r = idx / K0PAD, k = idx - r * K0PAD;
    float v = 0.0f;
    if (k < INDIM)             v = W_ih0[r * INDIM + k];
    else if (k < INDIM + PDIM) v = W_hh0[r * PDIM + (k - INDIM)];
    Wg0[idx] = f2bf(v);
    return;
  }
  idx -= S0;
  if (idx < S1) {
    int r = idx >> 8, k = idx & 255;
    float v = (k < PDIM) ? W_ih1[r * PDIM + k] : W_hh1[r * PDIM + (k - PDIM)];
    Wg1[idx] = f2bf(v);
    return;
  }
  idx -= S1;
  if (idx < S2) { Wr0[idx] = f2bf(W_hr0[idx]); return; }
  idx -= S2;
  if (idx < S2) { Wr1[idx] = f2bf(W_hr1[idx]); return; }
  idx -= S2;
  if (idx < 4096) { bb0[idx] = b_ih0[idx] + b_hh0[idx]; return; }
  idx -= 4096;
  if (idx < 4096) { bb1[idx] = b_ih1[idx] + b_hh1[idx]; }
}

extern "C" void kernel_launch(void* const* d_in, const int* in_sizes, int n_in,
                              void* d_out, int out_size, void* d_ws, size_t ws_size,
                              hipStream_t stream) {
  const float* input       = (const float*)d_in[0];
  const float* hidden      = (const float*)d_in[1];   // c-state [2,N,1024]
  const float* hidden_lstm = (const float*)d_in[2];   // h-state [2,N,128]
  const float* W_ih0 = (const float*)d_in[3];
  const float* W_hh0 = (const float*)d_in[4];
  const float* b_ih0 = (const float*)d_in[5];
  const float* b_hh0 = (const float*)d_in[6];
  const float* W_hr0 = (const float*)d_in[7];
  const float* W_ih1 = (const float*)d_in[8];
  const float* W_hh1 = (const float*)d_in[9];
  const float* b_ih1 = (const float*)d_in[10];
  const float* b_hh1 = (const float*)d_in[11];
  const float* W_hr1 = (const float*)d_in[12];
  const int*   curlen = (const int*)d_in[13];

  const int N = in_sizes[0] / INDIM;

  // d_ws layout (~5.8 MB total)
  unsigned short* Wg0 = (unsigned short*)d_ws;
  unsigned short* Wg1 = Wg0 + 4096 * K0PAD;
  unsigned short* Wr0 = Wg1 + 4096 * K1PAD;
  unsigned short* Wr1 = Wr0 + PDIM * HDIM;
  float* bb0 = (float*)(Wr1 + PDIM * HDIM);
  float* bb1 = bb0 + 4096;

  float* out = (float*)d_out;
  float* cn0 = out + (size_t)N * PDIM;
  float* cn1 = cn0 + (size_t)N * HDIM;
  float* hn0 = cn0 + (size_t)2 * N * HDIM;
  float* hn1 = hn0 + (size_t)N * PDIM;

  const int total = 4096 * K0PAD + 4096 * K1PAD + 2 * PDIM * HDIM + 2 * 4096;
  prep_weights<<<(total + 255) / 256, 256, 0, stream>>>(
      W_ih0, W_hh0, b_ih0, b_hh0, W_hr0,
      W_ih1, W_hh1, b_ih1, b_hh1, W_hr1,
      Wg0, Wg1, Wr0, Wr1, bb0, bb1);

  fused_lstm2_softmax<<<N / MROWS, 256, 0, stream>>>(
      input, hidden, hidden_lstm, curlen,
      Wg0, Wg1, Wr0, Wr1, bb0, bb1,
      out, cn0, cn1, hn0, hn1, N);
}
